// transformer_9826885173850
// MI455X (gfx1250) — matrix operand order repaired and checked by reading, not compiled, not run on hardware
//
#include <hip/hip_runtime.h>

// ---------------------------------------------------------------------------
// Transformer encoder-decoder forward for gfx1250 (MI455X), WMMA f16 GEMMs.
//
// Input flattening assumption (recursive dict-insertion order of setup_inputs):
//   d_in[0]=src(int32), d_in[1]=tar(int32), d_in[2]=emb_src, d_in[3]=emb_tgt,
//   d_in[4..99]   = 6 encoder layers x 16 arrays
//                   [g1,b1, wq,bq,wk,bk,wv,bv,wo,bo, g2,b2, w1,bb1,w2,bb2]
//   d_in[100..255]= 6 decoder layers x 26 arrays
//                   [g1,b1, self(wq..bo:8), g2,b2, cross(wq..bo:8), g3,b3,
//                    w1,bb1,w2,bb2]
//   d_in[256..261]= enc_g, enc_b, dec_g, dec_b, wout, bout
// ---------------------------------------------------------------------------

typedef __attribute__((ext_vector_type(16))) _Float16 v16h;
typedef __attribute__((ext_vector_type(8)))  _Float16 v8h;
typedef __attribute__((ext_vector_type(4)))  _Float16 v4h;
typedef __attribute__((ext_vector_type(8)))  float    v8f;

#define TDIV(a,b) (((a)+(b)-1)/(b))

// model constants
constexpr int Dm   = 512;
constexpr int Hh   = 8;
constexpr int DHh  = 64;
constexpr int DFFc = 2048;
constexpr int Vv   = 32000;
constexpr int Ll   = 6;
constexpr int Bb   = 4;
constexpr int Ss   = 512;
constexpr int BS   = Bb * Ss;     // 2048 token rows
constexpr float EPSc = 1e-5f;

// GEMM tiling: 256 threads = 8 wave32. Block tile 64(M) x 128(N) x 64(K).
// Each wave owns a 32x32 tile = 2x2 WMMA 16x16x32 f16 tiles, 2 k-substeps.
// Requirements (hold for every GEMM in this model): M%64==0, K%64==0, N%4==0.
constexpr int BM = 64, BN = 128, BK = 64;
constexpr int LDA_S = BK + 8;     // sA row stride in halves (144B, 16B-mult)
constexpr int LDB_S = BK + 8;     // sB is stored transposed: [n][k]

#define GF_RELU   1
#define GF_CAUSAL 2
#define GF_BT     4   // B operand is transposed in memory (N x K row-major)

// ---------------------------------------------------------------------------
// Generic fused GEMM: C = epilogue(alpha * A(MxK) @ B(KxN) + bias + res)
// mode 0: plain.
// mode 1: attention scores. z=(b*H+h). A=q head view, B=k head view (GF_BT);
//         C = scores + z*S*S.
// mode 2: attention A*V. A = probs + z*S*S; B = v head view; C = out head view.
// ---------------------------------------------------------------------------
__global__ __launch_bounds__(256)
void gemm_wmma_f16(const float* __restrict__ A, const float* __restrict__ Wm,
                   const float* __restrict__ bias, const float* __restrict__ res,
                   float* __restrict__ C,
                   int M, int N, int K, int lda, int ldw, int ldc,
                   float alpha, int flags, int mode)
{
    __shared__ _Float16 sA[BM * LDA_S];   // [m][k]
    __shared__ _Float16 sB[BN * LDB_S];   // [n][k]  (transposed tile)

    const int tid  = threadIdx.x;
    const int lane = tid & 31;
    const int wave = tid >> 5;
    const int lh   = lane & 15;
    const bool hiL = lane >= 16;
    const int row0 = blockIdx.y * BM;
    const int col0 = blockIdx.x * BN;
    const int z    = blockIdx.z;

    if (mode == 1) {
        const int b = z / Hh, h = z % Hh;
        const size_t o = (size_t)b * Ss * Dm + (size_t)h * DHh;
        A += o; Wm += o; C += (size_t)z * Ss * Ss;
    } else if (mode == 2) {
        const int b = z / Hh, h = z % Hh;
        const size_t o = (size_t)b * Ss * Dm + (size_t)h * DHh;
        A += (size_t)z * Ss * Ss; Wm += o; C += o;
    }

    const int wm0 = (wave >> 2) * 32;   // 0 / 32
    const int wn0 = (wave & 3) * 32;    // 0 / 32 / 64 / 96
    const bool bt = (flags & GF_BT) != 0;

    v8f acc[2][2];
#pragma unroll
    for (int i = 0; i < 2; ++i)
#pragma unroll
        for (int j = 0; j < 2; ++j)
#pragma unroll
            for (int r = 0; r < 8; ++r) acc[i][j][r] = 0.0f;

    for (int k0 = 0; k0 < K; k0 += BK) {
        // ---- stage A tile (BM x BK floats -> f16), guard-free, vectorized.
        // 1024 float4 vectors, 4 per thread.
        {
            float4 f[4];
#pragma unroll
            for (int j = 0; j < 4; ++j) {
                const int idx = tid + j * 256;
                const int r = idx >> 4, c4 = idx & 15;
                f[j] = *(const float4*)&A[(size_t)(row0 + r) * lda + k0 + c4 * 4];
            }
#pragma unroll
            for (int j = 0; j < 4; ++j) {
                const int idx = tid + j * 256;
                const int r = idx >> 4, c4 = idx & 15;
                v4h hv;
                hv[0] = (_Float16)f[j].x; hv[1] = (_Float16)f[j].y;
                hv[2] = (_Float16)f[j].z; hv[3] = (_Float16)f[j].w;
                *(v4h*)&sA[r * LDA_S + c4 * 4] = hv;
            }
        }
        // ---- stage B tile into transposed LDS layout [n][k].
        if (bt) {
            // Wt is [N][K] row-major: contiguous along k. 2048 vecs, 8/thread.
            float4 f[8];
#pragma unroll
            for (int j = 0; j < 8; ++j) {
                const int idx = tid + j * 256;
                const int n = idx >> 4, k4 = idx & 15;
                f[j] = (col0 + n < N)
                     ? *(const float4*)&Wm[(size_t)(col0 + n) * ldw + k0 + k4 * 4]
                     : float4{0.f, 0.f, 0.f, 0.f};
            }
#pragma unroll
            for (int j = 0; j < 8; ++j) {
                const int idx = tid + j * 256;
                const int n = idx >> 4, k4 = idx & 15;
                v4h hv;
                hv[0] = (_Float16)f[j].x; hv[1] = (_Float16)f[j].y;
                hv[2] = (_Float16)f[j].z; hv[3] = (_Float16)f[j].w;
                *(v4h*)&sB[n * LDB_S + k4 * 4] = hv;
            }
        } else {
            // W is [K][N] row-major: contiguous along n; scatter-transpose
            // into LDS. 2048 vecs, 8/thread.
            float4 f[8];
#pragma unroll
            for (int j = 0; j < 8; ++j) {
                const int idx = tid + j * 256;
                const int k = idx >> 5, n4 = idx & 31;
                f[j] = (col0 + n4 * 4 < N)
                     ? *(const float4*)&Wm[(size_t)(k0 + k) * ldw + col0 + n4 * 4]
                     : float4{0.f, 0.f, 0.f, 0.f};
            }
#pragma unroll
            for (int j = 0; j < 8; ++j) {
                const int idx = tid + j * 256;
                const int k = idx >> 5, n4 = idx & 31;
                sB[(n4 * 4 + 0) * LDB_S + k] = (_Float16)f[j].x;
                sB[(n4 * 4 + 1) * LDB_S + k] = (_Float16)f[j].y;
                sB[(n4 * 4 + 2) * LDB_S + k] = (_Float16)f[j].z;
                sB[(n4 * 4 + 3) * LDB_S + k] = (_Float16)f[j].w;
            }
        }
        // prefetch next K tile into cache (global_prefetch_b8)
        if (k0 + BK < K) {
            __builtin_prefetch(A + (size_t)(row0 + (tid & 63)) * lda + (k0 + BK), 0, 1);
            if (bt) {
                if (col0 + (tid & 127) < N)
                    __builtin_prefetch(Wm + (size_t)(col0 + (tid & 127)) * ldw + (k0 + BK), 0, 1);
            } else {
                __builtin_prefetch(Wm + (size_t)(k0 + BK + (tid & 63)) * ldw + col0, 0, 1);
            }
        }
        __syncthreads();

        // ---- two k=32 WMMA sub-steps over the staged 64-wide K tile.
#pragma unroll
        for (int kk = 0; kk < 2; ++kk) {
            const int kb = kk * 32;
            // A 16x32 f16 frag: lanes 0-15 (row M=lane): h[0..7]=K0..7,
            // h[8..15]=K16..23; lanes 16-31: K8..15 / K24..31.
            union Frag { v16h v; v8h h8[2]; _Float16 h[16]; };
            Frag af[2], bf[2];
#pragma unroll
            for (int mt = 0; mt < 2; ++mt) {
                const _Float16* p = &sA[(wm0 + mt * 16 + lh) * LDA_S + kb + (hiL ? 8 : 0)];
                af[mt].h8[0] = *(const v8h*)(p);
                af[mt].h8[1] = *(const v8h*)(p + 16);
            }
            // B 32x16 f16 frag: lane holds column n=lane%16; lanes 0-15:
            // h[0..7]=K0..7, h[8..15]=K16..23; lanes 16-31: K8..15 / K24..31.
            // sB is [n][k] -> two 16B LDS loads.
#pragma unroll
            for (int nt = 0; nt < 2; ++nt) {
                const _Float16* p = &sB[(wn0 + nt * 16 + lh) * LDB_S + kb + (hiL ? 8 : 0)];
                bf[nt].h8[0] = *(const v8h*)(p);
                bf[nt].h8[1] = *(const v8h*)(p + 16);
            }
#pragma unroll
            for (int mt = 0; mt < 2; ++mt)
#pragma unroll
                for (int nt = 0; nt < 2; ++nt)
                    acc[mt][nt] = __builtin_amdgcn_wmma_f32_16x16x32_f16(
                        false, af[mt].v, false, bf[nt].v,
                        (short)0, acc[mt][nt], false, false);
        }
        __syncthreads();
    }

    // Epilogue + store. C/D layout: VGPR r -> row (r + hi*8), col = lane%16.
#pragma unroll
    for (int mt = 0; mt < 2; ++mt) {
#pragma unroll
        for (int nt = 0; nt < 2; ++nt) {
            const int col = col0 + wn0 + nt * 16 + lh;
            if (col >= N) continue;
#pragma unroll
            for (int r = 0; r < 8; ++r) {
                const int row = row0 + wm0 + mt * 16 + r + (hiL ? 8 : 0);
                float v = acc[mt][nt][r] * alpha;
                if (bias) v += bias[col];
                if ((flags & GF_CAUSAL) && col > row) v = -1e9f;
                if (res) v += res[(size_t)row * ldc + col];
                if (flags & GF_RELU) v = fmaxf(v, 0.0f);
                C[(size_t)row * ldc + col] = v;
            }
        }
    }
}

// ---------------------------------------------------------------------------
// LayerNorm: one block per row.
// ---------------------------------------------------------------------------
__global__ __launch_bounds__(256)
void layernorm_k(const float* __restrict__ x, const float* __restrict__ g,
                 const float* __restrict__ b, float* __restrict__ o, int ncols)
{
    __shared__ float red[256];
    const int row = blockIdx.x;
    const float* xr = x + (size_t)row * ncols;

    float s = 0.0f;
    for (int c = threadIdx.x; c < ncols; c += 256) s += xr[c];
    red[threadIdx.x] = s; __syncthreads();
    for (int off = 128; off > 0; off >>= 1) {
        if (threadIdx.x < off) red[threadIdx.x] += red[threadIdx.x + off];
        __syncthreads();
    }
    const float mean = red[0] / (float)ncols;
    __syncthreads();

    float vs = 0.0f;
    for (int c = threadIdx.x; c < ncols; c += 256) {
        const float d = xr[c] - mean; vs += d * d;
    }
    red[threadIdx.x] = vs; __syncthreads();
    for (int off = 128; off > 0; off >>= 1) {
        if (threadIdx.x < off) red[threadIdx.x] += red[threadIdx.x + off];
        __syncthreads();
    }
    const float rstd = rsqrtf(red[0] / (float)ncols + EPSc);

    float* orow = o + (size_t)row * ncols;
    for (int c = threadIdx.x; c < ncols; c += 256)
        orow[c] = (xr[c] - mean) * rstd * g[c] + b[c];
}

// ---------------------------------------------------------------------------
// Row softmax in place: one block per row.
// ---------------------------------------------------------------------------
__global__ __launch_bounds__(256)
void softmax_k(float* __restrict__ s, int ncols)
{
    __shared__ float red[256];
    float* r = s + (size_t)blockIdx.x * ncols;

    float m = -3.4e38f;
    for (int c = threadIdx.x; c < ncols; c += 256) m = fmaxf(m, r[c]);
    red[threadIdx.x] = m; __syncthreads();
    for (int off = 128; off > 0; off >>= 1) {
        if (threadIdx.x < off)
            red[threadIdx.x] = fmaxf(red[threadIdx.x], red[threadIdx.x + off]);
        __syncthreads();
    }
    m = red[0]; __syncthreads();

    float sum = 0.0f;
    for (int c = threadIdx.x; c < ncols; c += 256) {
        const float e = __expf(r[c] - m);
        r[c] = e; sum += e;
    }
    red[threadIdx.x] = sum; __syncthreads();
    for (int off = 128; off > 0; off >>= 1) {
        if (threadIdx.x < off) red[threadIdx.x] += red[threadIdx.x + off];
        __syncthreads();
    }
    const float inv = 1.0f / red[0];
    for (int c = threadIdx.x; c < ncols; c += 256) r[c] *= inv;
}

// ---------------------------------------------------------------------------
// Embedding gather: one block per token, float4 copies.
// ---------------------------------------------------------------------------
__global__ __launch_bounds__(128)
void embed_k(const int* __restrict__ tok, const float* __restrict__ tab,
             float* __restrict__ o)
{
    const int t = blockIdx.x;
    const int id = tok[t];
    const float4* src = (const float4*)(tab + (size_t)id * Dm);
    float4* dst = (float4*)(o + (size_t)t * Dm);
    for (int i = threadIdx.x; i < Dm / 4; i += 128) dst[i] = src[i];
}

// ---------------------------------------------------------------------------
// Host-side orchestration.
// ---------------------------------------------------------------------------
static void gemm(hipStream_t st, const float* A, const float* W,
                 const float* bias, const float* res, float* C,
                 int M, int N, int K, int lda, int ldw, int ldc,
                 float alpha, int flags, int mode, int batches)
{
    dim3 grid(TDIV(N, BN), TDIV(M, BM), batches);
    gemm_wmma_f16<<<grid, 256, 0, st>>>(A, W, bias, res, C,
                                        M, N, K, lda, ldw, ldc,
                                        alpha, flags, mode);
}

extern "C" void kernel_launch(void* const* d_in, const int* in_sizes, int n_in,
                              void* d_out, int out_size, void* d_ws, size_t ws_size,
                              hipStream_t stream)
{
    (void)in_sizes; (void)n_in; (void)out_size;

    const int*   src     = (const int*)d_in[0];
    const int*   tar     = (const int*)d_in[1];
    const float* emb_src = (const float*)d_in[2];
    const float* emb_tgt = (const float*)d_in[3];
    auto P = [&](int i) { return (const float*)d_in[i]; };

    const int ENC0 = 4;
    const int DEC0 = ENC0 + Ll * 16;      // 100
    const int TAIL = DEC0 + Ll * 26;      // 256
    const float* enc_g = P(TAIL + 0); const float* enc_b = P(TAIL + 1);
    const float* dec_g = P(TAIL + 2); const float* dec_b = P(TAIL + 3);
    const float* wout  = P(TAIL + 4); const float* bout  = P(TAIL + 5);

    // Workspace partition (fp32):
    const size_t R = (size_t)BS * Dm;                 // 1M floats per activ buf
    float* x  = (float*)d_ws;                         // encoder stream
    float* y  = x  + R;                               // decoder stream
    float* nb = y  + R;                               // layernorm output
    float* q  = nb + R;
    float* k  = q  + R;
    float* v  = k  + R;
    float* t  = v  + R;                               // attention head-concat out
    float* eo = t  + R;                               // encoder final output
    float* hb = eo + R;                               // FFN hidden [BS, DFF]
    float* sc = hb + (size_t)BS * DFFc;               // scores [B*H, S, S]
    const size_t need = (8 * R + (size_t)BS * DFFc + (size_t)Bb * Hh * Ss * Ss)
                        * sizeof(float);              // ~80 MB
    if (ws_size < need) return;                       // workspace too small

    // Embeddings
    embed_k<<<BS, 128, 0, stream>>>(src, emb_src, x);
    embed_k<<<BS, 128, 0, stream>>>(tar, emb_tgt, y);

    // attention block: q from nb, k/v from kvsrc; io += proj(attn)
    auto attn = [&](float* io, const float* kvsrc, const float* const* w,
                    int causal)
    {
        gemm(stream, nb,    w[0], w[1], nullptr, q, BS, Dm, Dm, Dm, Dm, Dm,
             1.0f, 0, 0, 1);
        gemm(stream, kvsrc, w[2], w[3], nullptr, k, BS, Dm, Dm, Dm, Dm, Dm,
             1.0f, 0, 0, 1);
        gemm(stream, kvsrc, w[4], w[5], nullptr, v, BS, Dm, Dm, Dm, Dm, Dm,
             1.0f, 0, 0, 1);
        // scores[z] = (1/sqrt(DH)) * Qh @ Kh^T   (+ causal mask)
        gemm(stream, q, k, nullptr, nullptr, sc, Ss, Ss, DHh, Dm, Dm, Ss,
             0.125f, GF_BT | (causal ? GF_CAUSAL : 0), 1, Bb * Hh);
        softmax_k<<<Bb * Hh * Ss, 256, 0, stream>>>(sc, Ss);
        // out_h = probs @ Vh
        gemm(stream, sc, v, nullptr, nullptr, t, Ss, DHh, Ss, Ss, Dm, Dm,
             1.0f, 0, 2, Bb * Hh);
        // io = io + t @ wo + bo
        gemm(stream, t, w[6], w[7], io, io, BS, Dm, Dm, Dm, Dm, Dm,
             1.0f, 0, 0, 1);
    };

    // ---------------- Encoder ----------------
    for (int l = 0; l < Ll; ++l) {
        const float* lp[16];
        for (int i = 0; i < 16; ++i) lp[i] = P(ENC0 + l * 16 + i);
        layernorm_k<<<BS, 256, 0, stream>>>(x, lp[0], lp[1], nb, Dm);
        const float* aw[8] = {lp[2], lp[3], lp[4], lp[5], lp[6], lp[7], lp[8], lp[9]};
        attn(x, nb, aw, 0);
        layernorm_k<<<BS, 256, 0, stream>>>(x, lp[10], lp[11], nb, Dm);
        gemm(stream, nb, lp[12], lp[13], nullptr, hb, BS, DFFc, Dm,
             Dm, DFFc, DFFc, 1.0f, GF_RELU, 0, 1);
        gemm(stream, hb, lp[14], lp[15], x, x, BS, Dm, DFFc,
             DFFc, Dm, Dm, 1.0f, 0, 0, 1);
    }
    layernorm_k<<<BS, 256, 0, stream>>>(x, enc_g, enc_b, eo, Dm);

    // ---------------- Decoder ----------------
    for (int l = 0; l < Ll; ++l) {
        const float* lp[26];
        for (int i = 0; i < 26; ++i) lp[i] = P(DEC0 + l * 26 + i);
        // self-attention (causal)
        layernorm_k<<<BS, 256, 0, stream>>>(y, lp[0], lp[1], nb, Dm);
        const float* sw[8] = {lp[2], lp[3], lp[4], lp[5], lp[6], lp[7], lp[8], lp[9]};
        attn(y, nb, sw, 1);
        // cross-attention (k/v from encoder output)
        layernorm_k<<<BS, 256, 0, stream>>>(y, lp[10], lp[11], nb, Dm);
        const float* cw[8] = {lp[12], lp[13], lp[14], lp[15], lp[16], lp[17], lp[18], lp[19]};
        attn(y, eo, cw, 0);
        // FFN
        layernorm_k<<<BS, 256, 0, stream>>>(y, lp[20], lp[21], nb, Dm);
        gemm(stream, nb, lp[22], lp[23], nullptr, hb, BS, DFFc, Dm,
             Dm, DFFc, DFFc, 1.0f, GF_RELU, 0, 1);
        gemm(stream, hb, lp[24], lp[25], y, y, BS, Dm, DFFc,
             DFFc, Dm, Dm, 1.0f, 0, 0, 1);
    }
    layernorm_k<<<BS, 256, 0, stream>>>(y, dec_g, dec_b, nb, Dm);

    // logits = nb @ wout + bout   -> d_out [BS, V] fp32
    gemm(stream, nb, wout, bout, nullptr, (float*)d_out, BS, Vv, Dm,
         Dm, Vv, Vv, 1.0f, 0, 0, 1);
}
